// FightmLSTM_18399639896252
// MI455X (gfx1250) — compile-verified, matrix operand order checked by reading, not run on hardware
//
#include <hip/hip_runtime.h>
#include <hip/hip_bf16.h>

#define H_HEADS 4
#define D_HEAD 64
#define HID 256
#define T_SEQ 64
#define B_BATCH 256
#define BT (B_BATCH * T_SEQ)
#define IN_DIM 170
#define KP_IN 192

typedef __attribute__((ext_vector_type(16))) __bf16 v16bf;
typedef __attribute__((ext_vector_type(8)))  __bf16 v8bf;
typedef __attribute__((ext_vector_type(8)))  float  v8f;

__device__ __forceinline__ __bf16 f2bf(float f) {
  union { float f; unsigned u; } v; v.f = f;
  unsigned r = (v.u + 0x7FFFu + ((v.u >> 16) & 1u)) >> 16;
  union { unsigned short s; __bf16 b; } o; o.s = (unsigned short)r;
  return o.b;
}

// ---- f32 -> bf16 row convert with K padding (pad zeros) ----
__global__ void conv_pad_bf16(const float* __restrict__ in, __bf16* __restrict__ out,
                              int K, int Kpad) {
  int row = blockIdx.x;
  int t = threadIdx.x;                 // t in [0, Kpad)
  __bf16 v = (__bf16)0.0f;
  if (t < K) v = f2bf(in[(size_t)row * K + t]);
  out[(size_t)row * Kpad + t] = v;
}

// ---- transpose W (K x N, row-major f32) -> Wt (N x Kpad, bf16, zero-padded K) ----
__global__ void transpose_w_bf16(const float* __restrict__ W, __bf16* __restrict__ Wt,
                                 int K, int Kpad, int N) {
  int n = blockIdx.x;                  // [0, N)
  int k = threadIdx.x;                 // [0, Kpad)
  __bf16 v = (__bf16)0.0f;
  if (k < K) v = f2bf(W[(size_t)k * N + n]);
  Wt[(size_t)n * Kpad + k] = v;
}

// ---- build fused qkvo bias vector (cols 768..1023 = bo, rest zero) ----
__global__ void build_bias_qkvo(const float* __restrict__ bo, float* __restrict__ bias) {
  int n = threadIdx.x + blockIdx.x * blockDim.x;   // 1024
  bias[n] = (n >= 3 * HID) ? bo[n - 3 * HID] : 0.0f;
}

// ---- bf16 WMMA GEMM: C[M,N] = A[M,K] * Bt[N,K]^T + bias[N] ----
// One wave computes a 16x64 strip: one A fragment reused across 4 B fragments.
// K multiple of 32, N multiple of 64.
__global__ void __launch_bounds__(32)
gemm_bf16_wmma(const __bf16* __restrict__ A, const __bf16* __restrict__ Bt,
               const float* __restrict__ bias, float* __restrict__ C,
               int M, int N, int K) {
  int ntn = N >> 6;                    // 64-wide column strips
  int tile = blockIdx.x;
  int m0 = (tile / ntn) << 4;
  int n0 = (tile % ntn) << 6;
  int lane = threadIdx.x & 31;
  int r15 = lane & 15;
  int kb = (lane < 16) ? 0 : 8;        // 16-bit A/B VGPR layout: low lanes K0..7/16..23, high lanes K8..15/24..31
  const __bf16* arow = A + (size_t)(m0 + r15) * K + kb;
  const __bf16* brow = Bt + (size_t)(n0 + r15) * K + kb;
  size_t bstep = (size_t)16 * K;       // 16 columns of Bt
  v8f acc0 = {}, acc1 = {}, acc2 = {}, acc3 = {};
  for (int k0 = 0; k0 < K; k0 += 32) {
    if (k0 + 128 < K) {
      __builtin_prefetch(arow + k0 + 128, 0, 1);
      __builtin_prefetch(brow + k0 + 128, 0, 1);
    }
    v8bf alo = *(const v8bf*)(arow + k0);
    v8bf ahi = *(const v8bf*)(arow + k0 + 16);
    v16bf a;
#pragma unroll
    for (int i = 0; i < 8; ++i) { a[i] = alo[i]; a[8 + i] = ahi[i]; }
#pragma unroll
    for (int j = 0; j < 4; ++j) {
      const __bf16* bp = brow + (size_t)j * bstep + k0;
      v8bf blo = *(const v8bf*)(bp);
      v8bf bhi = *(const v8bf*)(bp + 16);
      v16bf b;
#pragma unroll
      for (int i = 0; i < 8; ++i) { b[i] = blo[i]; b[8 + i] = bhi[i]; }
      v8f& acc = (j == 0) ? acc0 : (j == 1) ? acc1 : (j == 2) ? acc2 : acc3;
      acc = __builtin_amdgcn_wmma_f32_16x16x32_bf16(false, a, false, b,
                                                    (short)0, acc, false, false);
    }
  }
  int mb = m0 + ((lane < 16) ? 0 : 8);
#pragma unroll
  for (int j = 0; j < 4; ++j) {
    int n = n0 + j * 16 + r15;
    float bv = bias ? bias[n] : 0.0f;
    v8f& acc = (j == 0) ? acc0 : (j == 1) ? acc1 : (j == 2) ? acc2 : acc3;
#pragma unroll
    for (int r = 0; r < 8; ++r)
      C[(size_t)(mb + r) * N + n] = acc[r] + bv;
  }
}

// ---- gate projections: i_raw/f_raw = h @ wi + bi, h @ wf + bf  (per b,t,head) ----
__global__ void gate_proj(const float* __restrict__ h, const float* __restrict__ wi,
                          const float* __restrict__ bi, const float* __restrict__ wf,
                          const float* __restrict__ bfv, float* __restrict__ ifr) {
  int row = blockIdx.x;                // [0, BT)
  int t = threadIdx.x;                 // 256
  __shared__ float red[256];
  float hv = h[(size_t)row * HID + t];
  for (int head = 0; head < H_HEADS; ++head) {
    red[t] = hv * wi[(size_t)t * H_HEADS + head];
    __syncthreads();
    for (int s = 128; s > 0; s >>= 1) { if (t < s) red[t] += red[t + s]; __syncthreads(); }
    if (t == 0) ifr[((size_t)row * H_HEADS + head) * 2 + 0] = red[0] + bi[head];
    __syncthreads();
    red[t] = hv * wf[(size_t)t * H_HEADS + head];
    __syncthreads();
    for (int s = 128; s > 0; s >>= 1) { if (t < s) red[t] += red[t + s]; __syncthreads(); }
    if (t == 0) ifr[((size_t)row * H_HEADS + head) * 2 + 1] = red[0] + bfv[head];
    __syncthreads();
  }
}

// ---- mLSTM recurrent scan: one block per (b, head), C[64x64] in registers ----
// q/k/v/o staged via CDNA5 async global->LDS DMA (ASYNCcnt).
__global__ void __launch_bounds__(256)
mlstm_scan(const float* __restrict__ qkvo, const float* __restrict__ ifr,
           float* __restrict__ out) {
  int bid = blockIdx.x;                // B*H
  int b = bid >> 2, head = bid & 3;
  int t = threadIdx.x;                 // 256
  int e = t & 63;                      // column of C
  int db = t >> 6;                     // row block (owns rows db*16 .. +15)

  __shared__ float stage[256];         // [0:64)=q raw, [64:128)=k, [128:192)=v, [192:256)=o raw
  __shared__ float nvec[64];
  __shared__ float part[4][64];
  __shared__ float nqred[64];
  __shared__ float gates[2];
  __shared__ float mstate;

  float Creg[16];
#pragma unroll
  for (int j = 0; j < 16; ++j) Creg[j] = 0.0f;
  if (t < 64) nvec[t] = 0.0f;
  if (t == 0) mstate = 0.0f;
  __syncthreads();

  for (int step = 0; step < T_SEQ; ++step) {
    size_t row = (size_t)b * T_SEQ + step;
    const float* base = qkvo + row * (4 * HID) + head * D_HEAD;
    // async-stage 4 x 64 floats (q,k,v,o chunks) into LDS: 64 lanes x 16B
    if (t < 64) {
      int c = t >> 4, i = t & 15;      // chunk, 16B slot
      const float* gsrc = base + c * HID + i * 4;
      unsigned lds_addr = (unsigned)(size_t)(&stage[c * 64 + i * 4]);
      unsigned long long gaddr = (unsigned long long)(size_t)gsrc;
      asm volatile("global_load_async_to_lds_b128 %0, %1, off"
                   :: "v"(lds_addr), "v"(gaddr) : "memory");
    }
    if (t == 0) {
      float it = ifr[(row * H_HEADS + head) * 2 + 0];
      float ft = ifr[(row * H_HEADS + head) * 2 + 1];
      float mo = mstate;
      float mt = fmaxf(ft + mo, it);
      gates[0] = __expf(ft + mo - mt);
      gates[1] = __expf(it - mt);
      mstate = mt;
    }
    asm volatile("s_wait_asynccnt 0x0" ::: "memory");
    __syncthreads();

    float fg = gates[0], ig = gates[1];
    float igk = ig * stage[64 + e];
    float cq = 0.0f;
#pragma unroll
    for (int j = 0; j < 16; ++j) {
      int d = db * 16 + j;
      Creg[j] = fg * Creg[j] + igk * stage[128 + d];
      cq += Creg[j] * stage[d];
    }
    part[db][e] = cq * 0.125f;         // fold q / sqrt(64)
    if (t < 64) {
      float nv = fg * nvec[t] + ig * stage[64 + t];
      nvec[t] = nv;
      nqred[t] = nv * stage[t] * 0.125f;
    }
    __syncthreads();
    if (t < 32) nqred[t] += nqred[t + 32]; __syncthreads();
    if (t < 16) nqred[t] += nqred[t + 16]; __syncthreads();
    if (t < 8)  nqred[t] += nqred[t + 8];  __syncthreads();
    if (t < 4)  nqred[t] += nqred[t + 4];  __syncthreads();
    if (t < 2)  nqred[t] += nqred[t + 2];  __syncthreads();
    if (t < 1)  nqred[t] += nqred[t + 1];  __syncthreads();

    if (t < 64) {
      float Cq = part[0][t] + part[1][t] + part[2][t] + part[3][t];
      float nq = fmaxf(fabsf(nqred[0]), 1.0f);
      float osig = 1.0f / (1.0f + __expf(-stage[192 + t]));
      out[row * HID + head * D_HEAD + t] = osig * (Cq / nq);
    }
    __syncthreads();
  }
}

// ---- LayerNorm over HID per (b,t) row ----
__global__ void layernorm_k(const float* __restrict__ in, const float* __restrict__ g,
                            const float* __restrict__ bta, float* __restrict__ out) {
  int row = blockIdx.x;
  int t = threadIdx.x;                 // 256
  __shared__ float red[256];
  __shared__ float mu, rstd;
  float v = in[(size_t)row * HID + t];
  red[t] = v; __syncthreads();
  for (int s = 128; s > 0; s >>= 1) { if (t < s) red[t] += red[t + s]; __syncthreads(); }
  if (t == 0) mu = red[0] * (1.0f / HID);
  __syncthreads();
  float c = v - mu;
  red[t] = c * c; __syncthreads();
  for (int s = 128; s > 0; s >>= 1) { if (t < s) red[t] += red[t + s]; __syncthreads(); }
  if (t == 0) rstd = rsqrtf(red[0] * (1.0f / HID) + 1e-5f);
  __syncthreads();
  out[(size_t)row * HID + t] = c * rstd * g[t] + bta[t];
}

// ---- head: z = relu(h[:, -1, :] @ W1 + b1) @ W2 + b2 ----
__global__ void head_mlp(const float* __restrict__ h, const float* __restrict__ W1,
                         const float* __restrict__ b1, const float* __restrict__ W2,
                         const float* __restrict__ b2, float* __restrict__ out) {
  int b = blockIdx.x;                  // 256
  int t = threadIdx.x;                 // 128
  const float* row = h + ((size_t)b * T_SEQ + (T_SEQ - 1)) * HID;
  float acc = b1[t];
  for (int k = 0; k < HID; ++k) acc += row[k] * W1[(size_t)k * 128 + t];
  acc = fmaxf(acc, 0.0f);
  __shared__ float red[128];
  for (int j = 0; j < 2; ++j) {
    red[t] = acc * W2[(size_t)t * 2 + j];
    __syncthreads();
    for (int s = 64; s > 0; s >>= 1) { if (t < s) red[t] += red[t + s]; __syncthreads(); }
    if (t == 0) out[b * 2 + j] = red[0] + b2[j];
    __syncthreads();
  }
}

extern "C" void kernel_launch(void* const* d_in, const int* in_sizes, int n_in,
                              void* d_out, int out_size, void* d_ws, size_t ws_size,
                              hipStream_t stream) {
  const float* x   = (const float*)d_in[0];
  const float* Wp  = (const float*)d_in[1];
  const float* bp  = (const float*)d_in[2];
  const float* Wq  = (const float*)d_in[3];
  const float* Wk  = (const float*)d_in[4];
  const float* Wv  = (const float*)d_in[5];
  const float* Wo  = (const float*)d_in[6];
  const float* bo  = (const float*)d_in[7];
  const float* wi  = (const float*)d_in[8];
  const float* bi  = (const float*)d_in[9];
  const float* wf  = (const float*)d_in[10];
  const float* bfv = (const float*)d_in[11];
  const float* lng = (const float*)d_in[12];
  const float* lnb = (const float*)d_in[13];
  const float* W1  = (const float*)d_in[14];
  const float* b1  = (const float*)d_in[15];
  const float* W2  = (const float*)d_in[16];
  const float* b2  = (const float*)d_in[17];
  float* out = (float*)d_out;

  char* ws = (char*)d_ws;
  size_t o = 0;
  auto alloc = [&](size_t bytes) { size_t r = o; o += (bytes + 255) & ~(size_t)255; return r; };

  __bf16* xbf   = (__bf16*)(ws + alloc((size_t)BT * KP_IN * 2));
  __bf16* hbf   = (__bf16*)(ws + alloc((size_t)BT * HID * 2));
  __bf16* WpT   = (__bf16*)(ws + alloc((size_t)HID * KP_IN * 2));
  __bf16* WfusT = (__bf16*)(ws + alloc((size_t)2 * 4 * HID * HID * 2)); // 2 layers x (1024 x 256)
  float*  biasQ = (float*)(ws + alloc((size_t)2 * 4 * HID * 4));
  float*  qkvo  = (float*)(ws + alloc((size_t)BT * 4 * HID * 4));
  float*  hA    = (float*)(ws + alloc((size_t)BT * HID * 4));
  float*  hB    = (float*)(ws + alloc((size_t)BT * HID * 4));
  float*  ifr   = (float*)(ws + alloc((size_t)BT * H_HEADS * 2 * 4));

  // 1. x -> bf16 padded to K=192
  conv_pad_bf16<<<BT, KP_IN, 0, stream>>>(x, xbf, IN_DIM, KP_IN);
  // 2. Wp^T -> bf16 (256 x 192)
  transpose_w_bf16<<<HID, KP_IN, 0, stream>>>(Wp, WpT, IN_DIM, KP_IN, HID);
  // 3. per-layer fused q|k|v|o weights, transposed (1024 rows x 256 K)
  for (int l = 0; l < 2; ++l) {
    const float* Ws[4] = { Wq + (size_t)l * HID * HID, Wk + (size_t)l * HID * HID,
                           Wv + (size_t)l * HID * HID, Wo + (size_t)l * HID * HID };
    for (int s = 0; s < 4; ++s)
      transpose_w_bf16<<<HID, HID, 0, stream>>>(Ws[s],
          WfusT + ((size_t)l * 4 + s) * HID * HID, HID, HID, HID);
    build_bias_qkvo<<<4, 256, 0, stream>>>(bo + (size_t)l * HID, biasQ + (size_t)l * 4 * HID);
  }
  // 4. input projection: hA = x @ Wp + bp   (M=BT, N=256, K=192)
  gemm_bf16_wmma<<<(BT / 16) * (HID / 64), 32, 0, stream>>>(xbf, WpT, bp, hA, BT, HID, KP_IN);

  for (int l = 0; l < 2; ++l) {
    // 5a. hA -> bf16
    conv_pad_bf16<<<BT, HID, 0, stream>>>(hA, hbf, HID, HID);
    // 5b. fused qkvo GEMM: (BT x 256) x (256 x 1024)
    gemm_bf16_wmma<<<(BT / 16) * (4 * HID / 64), 32, 0, stream>>>(
        hbf, WfusT + (size_t)l * 4 * HID * HID, biasQ + (size_t)l * 4 * HID,
        qkvo, BT, 4 * HID, HID);
    // 5c. gate projections
    gate_proj<<<BT, 256, 0, stream>>>(hA, wi + (size_t)l * HID * H_HEADS,
                                      bi + (size_t)l * H_HEADS,
                                      wf + (size_t)l * HID * H_HEADS,
                                      bfv + (size_t)l * H_HEADS, ifr);
    // 5d. recurrent scan (async LDS staging)
    mlstm_scan<<<B_BATCH * H_HEADS, 256, 0, stream>>>(qkvo, ifr, hB);
    // 5e. layernorm hB -> hA
    layernorm_k<<<BT, 256, 0, stream>>>(hB, lng + (size_t)l * HID, lnb + (size_t)l * HID, hA);
  }
  // 6. head MLP
  head_mlp<<<B_BATCH, 128, 0, stream>>>(hA, W1, b1, W2, b2, out);
}
